// GCNFN_16166256902433
// MI455X (gfx1250) — matrix operand order repaired and checked by reading, not compiled, not run on hardware
//
#include <hip/hip_runtime.h>
#include <hip/hip_bf16.h>

#define N_NODES 50000
#define N_EDGES 800000
#define N_GRAPH 128
#define F_IN    768
#define F_H     128
#define N_CLS   2
#define SLOPE   0.2f

typedef __attribute__((ext_vector_type(16))) __bf16 v16bf;
typedef __attribute__((ext_vector_type(8)))  __bf16 v8bf;
typedef __attribute__((ext_vector_type(4)))  __bf16 v4bf;
typedef __attribute__((ext_vector_type(8)))  float  v8f;
typedef __attribute__((ext_vector_type(4)))  float  v4f;
typedef int v4i_gcc __attribute__((vector_size(16)));   // matches builtin param type

#define AS1 __attribute__((address_space(1)))
#define AS3 __attribute__((address_space(3)))

#if defined(__has_builtin)
#if __has_builtin(__builtin_amdgcn_global_load_async_to_lds_b128)
#define HAVE_ASYNC_LDS 1
#endif
#if __has_builtin(__builtin_amdgcn_s_wait_asynccnt)
#define HAVE_WAIT_ASYNC 1
#endif
#endif

// 16-byte global -> LDS copy: async DMA (ASYNCcnt) when available, sync fallback.
__device__ __forceinline__ void copy16_g2l(void* lds_dst, const void* gsrc) {
#if defined(HAVE_ASYNC_LDS)
    __builtin_amdgcn_global_load_async_to_lds_b128(
        (AS1 v4i_gcc*)(uintptr_t)gsrc,
        (AS3 v4i_gcc*)(unsigned)(uintptr_t)lds_dst, 0, 0);
#else
    *(v4f*)lds_dst = *(const v4f*)gsrc;
#endif
}
__device__ __forceinline__ void wait_async_copies() {
#if defined(HAVE_ASYNC_LDS)
#if defined(HAVE_WAIT_ASYNC)
    __builtin_amdgcn_s_wait_asynccnt(0);
#else
    asm volatile("s_wait_asynccnt 0x0" ::: "memory");
#endif
#endif
}

__device__ __forceinline__ float selu_f(float x) {
    const float sc = 1.0507009873554805f, al = 1.6732632423543772f;
    return x > 0.f ? sc * x : sc * al * (__expf(x) - 1.f);
}
// order-preserving float<->uint encoding for atomic max over signed floats
__device__ __forceinline__ unsigned f2ord(float f) {
    unsigned u = __float_as_uint(f);
    return (u & 0x80000000u) ? ~u : (u | 0x80000000u);
}
__device__ __forceinline__ float ord2f(unsigned e) {
    unsigned u = (e & 0x80000000u) ? (e & 0x7FFFFFFFu) : ~e;
    return __uint_as_float(u);
}

// ---------------- fills ----------------
__global__ void fill_f32(float* p, float v, int n) {
    int i = blockIdx.x * blockDim.x + threadIdx.x;
    if (i < n) p[i] = v;
}
__global__ void fill_u32(unsigned* p, unsigned v, int n) {
    int i = blockIdx.x * blockDim.x + threadIdx.x;
    if (i < n) p[i] = v;
}

// ---- pre-pack W[K, F_H] fp32 -> bf16 in per-lane WMMA B-fragment layout ----
// Bp fragment index: ((kt*8 + colTile)*32 + lane)*16 + j
// lane: g = lane>>4, c = lane&15 ; element j -> B[k = kt*32 + g*8 + (j&7) + (j>=8 ? 16 : 0)][colTile*16 + c]
__global__ void prepack_w_bf16(const float* __restrict__ W, __bf16* __restrict__ Wp, int K) {
    int id = blockIdx.x * blockDim.x + threadIdx.x;      // over K*F_H
    if (id >= K * F_H) return;
    int j    = id & 15;
    int lane = (id >> 4) & 31;
    int t    = id >> 9;                                  // kt*8 + ct
    int kt = t >> 3, ct = t & 7;
    int g = lane >> 4, cc = lane & 15;
    int k   = kt * 32 + g * 8 + (j & 7) + ((j >> 3) << 4);
    int col = ct * 16 + cc;
    Wp[id] = (__bf16)W[(long)k * F_H + col];
}

// ---------------- WMMA GEMM: Z[N,128] = X[N,K] @ W[K,128] ----------------
// block = 256 thr (8 waves). Block computes 64 rows x 128 cols; wave w owns col
// tile w and 4 row sub-tiles (4 wmma per k-step off one B fragment).
// Pipeline per k-step: async DMA fp32 tile -> fT (double buffered), one
// cooperative fp32->bf16 pass writes bT in per-lane FRAGMENT order, then each
// wave does one contiguous 32B ds load per fragment + wmma (no VALU cvt).
#define MT 64
__global__ void __launch_bounds__(256)
gat_gemm_wmma(const float* __restrict__ X, const v16bf* __restrict__ Bp,
              float* __restrict__ Z, int K) {
    __shared__ float  fT[2][MT * 32];          // raw fp32 tiles  (16 KB)
    __shared__ __bf16 bT[2][4 * 32 * 16];      // fragment-order bf16 (8 KB)
    const int tid  = threadIdx.x;
    const int lane = tid & 31;
    const int wave = tid >> 5;                 // col tile 0..7
    const int rowBase = blockIdx.x * MT;
    const int ksteps = K >> 5;

    v8f acc[4] = {};

    // staging coords: 512 float4 quads per tile, 2 per thread
    // quad q: row = q>>3, col = (q&7)*4
    auto issue = [&](int kt, int buf) {
        #pragma unroll
        for (int p = 0; p < 2; ++p) {
            int q   = tid + p * 256;
            int row = q >> 3;
            int col = (q & 7) * 4;
            int grow = rowBase + row;
            grow = grow < N_NODES ? grow : N_NODES - 1;   // clamp, no divergence
            copy16_g2l(&fT[buf][row * 32 + col],
                       X + (long)grow * K + (kt << 5) + col);
        }
    };

    issue(0, 0);
    for (int kt = 0; kt < ksteps; ++kt) {
        const int cur = kt & 1;
        wait_async_copies();
        __syncthreads();                          // fT[cur] resident
        if (kt + 1 < ksteps) issue(kt + 1, cur ^ 1);

        // cooperative convert: fT[cur] -> bT[cur] in fragment order.
        // fragment element j of (subtile m, lane): g=lane>>4, rr=lane&15:
        //   j<8 -> k = g*8 + j ; j>=8 -> k = 16 + g*8 + (j-8), row = m*16+rr
        #pragma unroll
        for (int p = 0; p < 2; ++p) {
            int q   = tid + p * 256;
            int row = q >> 3;                     // 0..63
            int col = (q & 7) * 4;                // 0,4,..,28
            v4f f = *(const v4f*)&fT[cur][row * 32 + col];
            v4bf h;
            #pragma unroll
            for (int i = 0; i < 4; ++i) h[i] = (__bf16)f[i];
            int m  = row >> 4, rr = row & 15;
            int lane2 = rr + (((col >> 3) & 1) ? 16 : 0);
            int ebase = (col & 7) + (col >= 16 ? 8 : 0);
            *(v4bf*)&bT[cur][(m * 32 + lane2) * 16 + ebase] = h;
        }
        __syncthreads();                          // bT[cur] ready

        // B fragment: one 32B load, reused by 4 wmma
        v16bf b = Bp[(kt * 8 + wave) * 32 + lane];
        const v16bf* frag = (const v16bf*)&bT[cur][0];
        #pragma unroll
        for (int m = 0; m < 4; ++m) {
            v16bf a = frag[m * 32 + lane];        // contiguous 32B ds load
            acc[m] = __builtin_amdgcn_wmma_f32_16x16x32_bf16(false, a, false, b,
                                                             (short)0, acc[m],
                                                             false, false);
        }
    }

    // C/D layout: lane n<16 holds rows 0..7 at col n; lane n+16 rows 8..15
    const int col = wave * 16 + (lane & 15);
    const int ro  = (lane >> 4) * 8;
    #pragma unroll
    for (int m = 0; m < 4; ++m) {
        #pragma unroll
        for (int i = 0; i < 8; ++i) {
            int row = rowBase + m * 16 + ro + i;
            if (row < N_NODES) Z[(long)row * F_H + col] = acc[m][i];
        }
    }
}

// --------------- per-node attention scores: s = z . a ---------------
__global__ void __launch_bounds__(256)
node_scores(const float* __restrict__ Z, const float* __restrict__ a_src,
            const float* __restrict__ a_dst, float* __restrict__ s_src,
            float* __restrict__ s_dst, int n) {
    int lane = threadIdx.x & 31;
    int node = blockIdx.x * 8 + (threadIdx.x >> 5);
    if (node >= n) return;
    float as = 0.f, ad = 0.f;
    #pragma unroll
    for (int i = 0; i < 4; ++i) {
        int d = lane + i * 32;
        float z = Z[(long)node * F_H + d];
        as += z * a_src[d];
        ad += z * a_dst[d];
    }
    #pragma unroll
    for (int off = 16; off; off >>= 1) {
        as += __shfl_down(as, off);
        ad += __shfl_down(ad, off);
    }
    if (lane == 0) { s_src[node] = as; s_dst[node] = ad; }
}

// --------------- edge softmax phases ---------------
__global__ void edge_max(const int* __restrict__ ei, const float* __restrict__ s_src,
                         const float* __restrict__ s_dst, unsigned* __restrict__ m_enc) {
    int e = blockIdx.x * blockDim.x + threadIdx.x;
    if (e >= N_EDGES) return;
    int s = ei[e], d = ei[N_EDGES + e];
    float v = s_src[s] + s_dst[d];
    v = v > 0.f ? v : SLOPE * v;
    atomicMax(&m_enc[d], f2ord(v));
}

__global__ void edge_expsum(const int* __restrict__ ei, const float* __restrict__ s_src,
                            const float* __restrict__ s_dst, const unsigned* __restrict__ m_enc,
                            float* __restrict__ ex, float* __restrict__ denom) {
    int e = blockIdx.x * blockDim.x + threadIdx.x;
    if (e >= N_EDGES) return;
    int s = ei[e], d = ei[N_EDGES + e];
    float v = s_src[s] + s_dst[d];
    v = v > 0.f ? v : SLOPE * v;
    float x = __expf(v - ord2f(m_enc[d]));
    ex[e] = x;
    atomicAdd(&denom[d], x);
}

__global__ void edge_scatter(const int* __restrict__ ei, const float* __restrict__ ex,
                             const float* __restrict__ denom, const float* __restrict__ Z,
                             float* __restrict__ Hout) {
    int id = blockIdx.x * blockDim.x + threadIdx.x;   // over E * F_H (fits int)
    int e = id >> 7;                                  // / 128
    if (e >= N_EDGES) return;
    int d  = id & 127;
    int sn = ei[e], dn = ei[N_EDGES + e];
    float alpha = ex[e] / (denom[dn] + 1e-9f);
    atomicAdd(&Hout[(long)dn * F_H + d], alpha * Z[(long)sn * F_H + d]);
}

__global__ void bias_selu(float* __restrict__ Hbuf, const float* __restrict__ b, int total) {
    int i = blockIdx.x * blockDim.x + threadIdx.x;
    if (i >= total) return;
    Hbuf[i] = selu_f(Hbuf[i] + b[i & 127]);
}

// --------------- pooling ---------------
__global__ void pool_sum(const float* __restrict__ Hbuf, const int* __restrict__ batch,
                         float* __restrict__ pooled) {
    int i = blockIdx.x * blockDim.x + threadIdx.x;    // N * F_H
    int n = i >> 7;
    if (n >= N_NODES) return;
    int d = i & 127;
    atomicAdd(&pooled[(long)batch[n] * F_H + d], Hbuf[i]);
}
__global__ void pool_cnt(const int* __restrict__ batch, float* __restrict__ cnt) {
    int n = blockIdx.x * blockDim.x + threadIdx.x;
    if (n >= N_NODES) return;
    atomicAdd(&cnt[batch[n]], 1.f);
}

// --------------- MLP head + log_softmax (tiny: G=128) ---------------
__global__ void __launch_bounds__(128)
mlp_head(const float* __restrict__ pooled, const float* __restrict__ cnt,
         const float* __restrict__ fc1w, const float* __restrict__ fc1b,
         const float* __restrict__ fc2w, const float* __restrict__ fc2b,
         float* __restrict__ out) {
    __shared__ float sp[F_H];
    __shared__ float so[F_H];
    __shared__ float lg[N_CLS];
    int g = blockIdx.x, j = threadIdx.x;
    float c = cnt[g]; c = c < 1.f ? 1.f : c;
    sp[j] = pooled[(long)g * F_H + j] / c;
    __syncthreads();
    float acc = fc1b[j];
    #pragma unroll 8
    for (int d = 0; d < F_H; ++d) acc += sp[d] * fc1w[d * F_H + j];
    so[j] = selu_f(acc);
    __syncthreads();
    if (j < N_CLS) {
        float l = fc2b[j];
        for (int d = 0; d < F_H; ++d) l += so[d] * fc2w[d * N_CLS + j];
        lg[j] = l;
    }
    __syncthreads();
    if (j == 0) {
        float m = lg[0] > lg[1] ? lg[0] : lg[1];
        float lse = m + __logf(__expf(lg[0] - m) + __expf(lg[1] - m));
        out[g * N_CLS + 0] = lg[0] - lse;
        out[g * N_CLS + 1] = lg[1] - lse;
    }
}

static inline int cdiv(long a, long b) { return (int)((a + b - 1) / b); }

extern "C" void kernel_launch(void* const* d_in, const int* in_sizes, int n_in,
                              void* d_out, int out_size, void* d_ws, size_t ws_size,
                              hipStream_t stream) {
    const float* x      = (const float*)d_in[0];
    const int*   eidx   = (const int*)  d_in[1];
    const int*   batch  = (const int*)  d_in[2];
    const float* W1     = (const float*)d_in[3];
    const float* a1s    = (const float*)d_in[4];
    const float* a1d    = (const float*)d_in[5];
    const float* b1     = (const float*)d_in[6];
    const float* W2     = (const float*)d_in[7];
    const float* a2s    = (const float*)d_in[8];
    const float* a2d    = (const float*)d_in[9];
    const float* b2     = (const float*)d_in[10];
    const float* fc1w   = (const float*)d_in[11];
    const float* fc1b   = (const float*)d_in[12];
    const float* fc2w   = (const float*)d_in[13];
    const float* fc2b   = (const float*)d_in[14];
    float* out = (float*)d_out;

    // workspace layout (all offsets 32B aligned)
    float*    buf0   = (float*)d_ws;                          // z   [N,H]
    float*    buf1   = buf0 + (size_t)N_NODES * F_H;          // h   [N,H]
    float*    s_src  = buf1 + (size_t)N_NODES * F_H;          // [N]
    float*    s_dst  = s_src + N_NODES;                       // [N]
    unsigned* m_enc  = (unsigned*)(s_dst + N_NODES);          // [N]
    float*    denom  = (float*)(m_enc + N_NODES);             // [N]
    float*    ex     = denom + N_NODES;                       // [E]
    float*    pooled = ex + N_EDGES;                          // [G,H]
    float*    cnt    = pooled + (size_t)N_GRAPH * F_H;        // [G]
    __bf16*   Wp1    = (__bf16*)(cnt + N_GRAPH);              // IN*H bf16
    __bf16*   Wp2    = Wp1 + (size_t)F_IN * F_H;              // H*H bf16

    const unsigned NEG_INF_ENC = 0x007FFFFFu;  // f2ord(-inf)
    const int NH = N_NODES * F_H;

    // pre-pack weights into WMMA B-fragment bf16 layout
    prepack_w_bf16<<<cdiv((long)F_IN * F_H, 256), 256, 0, stream>>>(W1, Wp1, F_IN);
    prepack_w_bf16<<<cdiv((long)F_H  * F_H, 256), 256, 0, stream>>>(W2, Wp2, F_H);

    // ---------------- GAT layer (shared sequence) ----------------
    auto run_layer = [&](const float* Xin, int K, const __bf16* Wp,
                         const float* as, const float* ad, const float* bias) {
        gat_gemm_wmma<<<cdiv(N_NODES, MT), 256, 0, stream>>>(Xin, (const v16bf*)Wp, buf0, K);
        node_scores<<<N_NODES / 8, 256, 0, stream>>>(buf0, as, ad, s_src, s_dst, N_NODES);
        fill_u32<<<cdiv(N_NODES, 256), 256, 0, stream>>>(m_enc, NEG_INF_ENC, N_NODES);
        fill_f32<<<cdiv(N_NODES, 256), 256, 0, stream>>>(denom, 0.f, N_NODES);
        fill_f32<<<cdiv(NH, 256), 256, 0, stream>>>(buf1, 0.f, NH);
        edge_max<<<cdiv(N_EDGES, 256), 256, 0, stream>>>(eidx, s_src, s_dst, m_enc);
        edge_expsum<<<cdiv(N_EDGES, 256), 256, 0, stream>>>(eidx, s_src, s_dst, m_enc, ex, denom);
        edge_scatter<<<cdiv((long)N_EDGES * F_H, 256), 256, 0, stream>>>(eidx, ex, denom, buf0, buf1);
        bias_selu<<<cdiv(NH, 256), 256, 0, stream>>>(buf1, bias, NH);
    };

    run_layer(x,    F_IN, Wp1, a1s, a1d, b1);   // layer 1: x -> buf1
    run_layer(buf1, F_H,  Wp2, a2s, a2d, b2);   // layer 2: buf1 -> buf1 (via buf0)

    // ---------------- pooling + head ----------------
    fill_f32<<<cdiv(N_GRAPH * F_H, 256), 256, 0, stream>>>(pooled, 0.f, N_GRAPH * F_H);
    fill_f32<<<1, 256, 0, stream>>>(cnt, 0.f, N_GRAPH);
    pool_sum<<<cdiv(NH, 256), 256, 0, stream>>>(buf1, batch, pooled);
    pool_cnt<<<cdiv(N_NODES, 256), 256, 0, stream>>>(batch, cnt);
    mlp_head<<<N_GRAPH, 128, 0, stream>>>(pooled, cnt, fc1w, fc1b, fc2w, fc2b, out);

    (void)in_sizes; (void)n_in; (void)out_size; (void)ws_size;
}
